// BorderHead_21234318311587
// MI455X (gfx1250) — matrix operand order repaired and checked by reading, not compiled, not run on hardware
//
#include <hip/hip_runtime.h>
#include <hip/hip_bf16.h>
#include <math.h>

// ---------------------------------------------------------------------------
// BorderDet head on MI455X (gfx1250): all convs lowered to implicit GEMM on
// v_wmma_f32_16x16x32_f16 (f16 in / f32 accum; GN/IN renormalizes each layer).
// Activations NHWC f16; 3x3 convs use a tap-loop im2col.  Wave tile =
// 64(M) x 32(N): 8 WMMA per 32-wide K-chunk.  Channel loop is branchless
// (zero-page pointer select for pad taps, clamped second B row), and all
// fragment pointers are address-space(1) so loads lower to global_load_b128
// (LOADcnt only) instead of flat_load (LOADcnt+DScnt).
// ---------------------------------------------------------------------------

typedef _Float16 h16;
typedef _Float16 v16h __attribute__((ext_vector_type(16)));
typedef _Float16 v8h  __attribute__((ext_vector_type(8)));
typedef float    v8f  __attribute__((ext_vector_type(8)));

// Global (address-space 1) pointer types: force global_load_* codegen even
// through per-lane pointer selects.
typedef const h16 __attribute__((address_space(1)))* gch;
typedef const v8h __attribute__((address_space(1)))* gcv8;

__device__ __forceinline__ gch to_global(const h16* p) {
  return (gch)(unsigned long long)p;
}

#define GN_EPS 1e-5f

// Build a 16-half fragment from two 16B global loads `gap` halfs apart.
__device__ __forceinline__ v16h ld_frag(gch p, int gap) {
  union { v16h v; v8h h[2]; } u;
  u.h[0] = *(gcv8)p;
  u.h[1] = *(gcv8)(p + gap);
  return u.v;
}

// ---- zero fill (for the pad zero-page) -------------------------------------
__global__ void zero_h16(h16* __restrict__ p, long n) {
  long i = (long)blockIdx.x * blockDim.x + threadIdx.x;
  if (i < n) p[i] = (h16)0.f;
}

// ---- weight pack: OIHW f32 -> Bt f16 [Npad][K], k = tap*Cin + c ------------
__global__ void pack_w(const float* __restrict__ w, h16* __restrict__ bt,
                       int O, int Npad, int Cin, int khw) {
  long K = (long)khw * Cin;
  long i = (long)blockIdx.x * blockDim.x + threadIdx.x;
  if (i >= (long)Npad * K) return;
  int o = (int)(i / K);
  int k = (int)(i % K);
  int tap = k / Cin, c = k - tap * Cin;
  float v = (o < O) ? w[((long)(o * Cin + c)) * khw + tap] : 0.f;
  bt[i] = (h16)v;
}

// ---- NCHW f32 -> NHWC f16 --------------------------------------------------
__global__ void nchw_to_nhwc_h(const float* __restrict__ in, h16* __restrict__ out,
                               int C, int HW) {
  long i = (long)blockIdx.x * blockDim.x + threadIdx.x;
  if (i >= (long)C * HW) return;
  int c = (int)(i / HW), p = (int)(i % HW);
  out[(long)p * C + c] = (h16)in[i];
}

// ---- WMMA implicit-GEMM ----------------------------------------------------
// epi: 0 = f32 NHWC raw, 1 = f32 NCHW raw, 2 = f32 NCHW relu(y*alpha)*beta,
//      3 = f16 NHWC relu.   taps = 9 (3x3 pad=1 im2col) or 1 (1x1).
__global__ __launch_bounds__(128) void gemm_wmma(
    const h16* __restrict__ A, const h16* __restrict__ Bt,
    const h16* __restrict__ zpage,
    const float* __restrict__ bias, const float* __restrict__ alpha_ptr, float beta,
    float* __restrict__ outF, h16* __restrict__ outH,
    int M, int Npad, int Nreal, int taps, int H, int W, int Cin, int epi) {
  const int lane = threadIdx.x;
  const int r = lane & 15;
  const int hiHalf = lane >> 4;            // 0: lanes 0-15, 1: lanes 16-31
  const int m0 = blockIdx.x * 64;
  const int n0 = (blockIdx.y * 4 + threadIdx.y) * 32;
  if (n0 >= Npad) return;
  const bool hasN1 = (n0 + 16) < Npad;         // wave-uniform
  const int n1base = hasN1 ? (n0 + 16) : n0;   // clamp: no branch in loop

  const int K = taps * Cin;
  const int kbA = hiHalf * 8;              // A frag: K {kb..kb+7, kb+16..kb+23}
  const int kbB = hiHalf * 8;              // B frag: K {kb..kb+7, kb+16..kb+23}
  const gch Ag = to_global(A);
  const gch Zg = to_global(zpage);
  const gch brow0 = to_global(Bt) + (long)(n0 + r) * K + kbB;
  const gch brow1 = to_global(Bt) + (long)(n1base + r) * K + kbB;
  const h16* brow0_gen = Bt + (long)(n0 + r) * K + kbB;  // for prefetch only

  // pixel coords per m-tile (hoisted divisions; 4 total)
  int py[4], px[4];
#pragma unroll
  for (int t = 0; t < 4; ++t) {
    int m = m0 + t * 16 + r;
    py[t] = m / W;
    px[t] = m - py[t] * W;
  }

  v8f acc[4][2] = {};
  for (int tap = 0; tap < taps; ++tap) {
    // Per-tap A pointers: invalid taps redirected to the zero page via a
    // per-lane pointer select (v_cndmask) -- channel loop stays branchless.
    gch ap[4];
    if (taps == 9) {
      int dy = tap / 3 - 1;
      int dx = tap - (tap / 3) * 3 - 1;
#pragma unroll
      for (int t = 0; t < 4; ++t) {
        int sy = py[t] + dy, sx = px[t] + dx;
        bool ok = (sy >= 0) && (sy < H) && (sx >= 0) && (sx < W);
        gch base = ok ? (Ag + (long)(sy * W + sx) * Cin) : Zg;
        ap[t] = base + kbA;
      }
    } else {
#pragma unroll
      for (int t = 0; t < 4; ++t)
        ap[t] = Ag + (long)(m0 + t * 16 + r) * Cin + kbA;
    }
    const gch b0p = brow0 + (long)tap * Cin;
    const gch b1p = brow1 + (long)tap * Cin;
    const h16* bpre = brow0_gen + (long)tap * Cin;

#pragma unroll 2
    for (int c0 = 0; c0 < Cin; c0 += 32) {
      v16h b0 = ld_frag(b0p + c0, 16);
      v16h b1 = ld_frag(b1p + c0, 16);
      __builtin_prefetch(bpre + c0 + 32, 0, 3);  // global_prefetch_b8 (speculative)
      v16h a0 = ld_frag(ap[0] + c0, 16);
      v16h a1 = ld_frag(ap[1] + c0, 16);
      v16h a2 = ld_frag(ap[2] + c0, 16);
      v16h a3 = ld_frag(ap[3] + c0, 16);
      acc[0][0] = __builtin_amdgcn_wmma_f32_16x16x32_f16(false, a0, false, b0, (short)0, acc[0][0], false, false);
      acc[0][1] = __builtin_amdgcn_wmma_f32_16x16x32_f16(false, a0, false, b1, (short)0, acc[0][1], false, false);
      acc[1][0] = __builtin_amdgcn_wmma_f32_16x16x32_f16(false, a1, false, b0, (short)0, acc[1][0], false, false);
      acc[1][1] = __builtin_amdgcn_wmma_f32_16x16x32_f16(false, a1, false, b1, (short)0, acc[1][1], false, false);
      acc[2][0] = __builtin_amdgcn_wmma_f32_16x16x32_f16(false, a2, false, b0, (short)0, acc[2][0], false, false);
      acc[2][1] = __builtin_amdgcn_wmma_f32_16x16x32_f16(false, a2, false, b1, (short)0, acc[2][1], false, false);
      acc[3][0] = __builtin_amdgcn_wmma_f32_16x16x32_f16(false, a3, false, b0, (short)0, acc[3][0], false, false);
      acc[3][1] = __builtin_amdgcn_wmma_f32_16x16x32_f16(false, a3, false, b1, (short)0, acc[3][1], false, false);
    }
  }

  const float alpha = alpha_ptr ? alpha_ptr[0] : 1.f;
#pragma unroll
  for (int j = 0; j < 2; ++j) {
    if (j == 1 && !hasN1) break;
    const int n = n0 + j * 16 + r;
    if (n >= Nreal) continue;
    const float bv = bias ? bias[n] : 0.f;
#pragma unroll
    for (int t = 0; t < 4; ++t) {
#pragma unroll
      for (int i = 0; i < 8; ++i) {
        int m = m0 + t * 16 + i + hiHalf * 8;  // C layout: VGPR i, half-wave
        float y = acc[t][j][i] + bv;
        if (epi == 0) {
          outF[(long)m * Nreal + n] = y;
        } else if (epi == 1) {
          outF[(long)n * M + m] = y;
        } else if (epi == 2) {
          float u = y * alpha;
          u = u > 0.f ? u : 0.f;
          outF[(long)n * M + m] = u * beta;
        } else {
          outH[(long)m * Nreal + n] = (h16)(y > 0.f ? y : 0.f);
        }
      }
    }
  }
}

// ---- GroupNorm (32 groups, C=256): stats + apply+relu ----------------------
__global__ __launch_bounds__(256) void gn_stats(const float* __restrict__ x,
                                                float* __restrict__ st, int HW) {
  __shared__ float s1[256], s2[256];
  const int g = blockIdx.x, t = threadIdx.x;
  float a = 0.f, b = 0.f;
  const int tot = HW * 8;
  for (int i = t; i < tot; i += 256) {
    int m = i >> 3, c = (g << 3) + (i & 7);
    float v = x[(long)m * 256 + c];
    a += v; b += v * v;
  }
  s1[t] = a; s2[t] = b;
  __syncthreads();
  for (int o = 128; o > 0; o >>= 1) {
    if (t < o) { s1[t] += s1[t + o]; s2[t] += s2[t + o]; }
    __syncthreads();
  }
  if (t == 0) {
    float inv = 1.f / (float)tot;
    float mean = s1[0] * inv;
    float var = s2[0] * inv - mean * mean;
    st[g * 2] = mean;
    st[g * 2 + 1] = rsqrtf(var + GN_EPS);
  }
}

__global__ void gn_apply(const float* __restrict__ x, const float* __restrict__ st,
                         const float* __restrict__ gam, const float* __restrict__ bet,
                         h16* __restrict__ out, int HW) {
  long i = (long)blockIdx.x * blockDim.x + threadIdx.x;
  if (i >= (long)HW * 256) return;
  int c = (int)(i & 255);
  int g = c >> 3;
  float y = (x[i] - st[g * 2]) * st[g * 2 + 1] * gam[c] + bet[c];
  out[i] = (h16)(y > 0.f ? y : 0.f);
}

// ---- InstanceNorm (per channel over HW): stats + apply+relu ----------------
__global__ __launch_bounds__(256) void in_stats(const float* __restrict__ x,
                                                float* __restrict__ st, int C, int HW) {
  __shared__ float s1[256], s2[256];
  const int c = blockIdx.x, t = threadIdx.x;
  float a = 0.f, b = 0.f;
  for (int m = t; m < HW; m += 256) {
    float v = x[(long)m * C + c];
    a += v; b += v * v;
  }
  s1[t] = a; s2[t] = b;
  __syncthreads();
  for (int o = 128; o > 0; o >>= 1) {
    if (t < o) { s1[t] += s1[t + o]; s2[t] += s2[t + o]; }
    __syncthreads();
  }
  if (t == 0) {
    float inv = 1.f / (float)HW;
    float mean = s1[0] * inv;
    float var = s2[0] * inv - mean * mean;
    st[c * 2] = mean;
    st[c * 2 + 1] = rsqrtf(var + GN_EPS);
  }
}

__global__ void in_apply(const float* __restrict__ x, const float* __restrict__ st,
                         h16* __restrict__ out, int C, int HW) {
  long i = (long)blockIdx.x * blockDim.x + threadIdx.x;
  if (i >= (long)HW * C) return;
  int c = (int)(i % C);
  float y = (x[i] - st[c * 2]) * st[c * 2 + 1];
  out[i] = (h16)(y > 0.f ? y : 0.f);
}

// ---- pre_bbox + clipped boxes from bbox_reg (NCHW) + shifts ----------------
__global__ void make_boxes(const float* __restrict__ bp, const float* __restrict__ sh,
                           float* __restrict__ pre, float* __restrict__ boxes,
                           int HW, int H, int W, float s) {
  int p = blockIdx.x * blockDim.x + threadIdx.x;
  if (p >= HW) return;
  float o0 = bp[p], o1 = bp[HW + p], o2 = bp[2 * HW + p], o3 = bp[3 * HW + p];
  float sx = sh[p * 2], sy = sh[p * 2 + 1];
  float p0 = sx - o0, p1 = sy - o1, p2 = sx + o2, p3 = sy + o3;
  pre[p * 4 + 0] = p0; pre[p * 4 + 1] = p1;
  pre[p * 4 + 2] = p2; pre[p * 4 + 3] = p3;
  float inv = 1.f / s;
  boxes[p * 4 + 0] = fminf(fmaxf(p0 * inv, 0.f), (float)(W - 1));
  boxes[p * 4 + 1] = fminf(fmaxf(p1 * inv, 0.f), (float)(H - 1));
  boxes[p * 4 + 2] = fminf(fmaxf(p2 * inv, 0.f), (float)(W - 1));
  boxes[p * 4 + 3] = fminf(fmaxf(p3 * inv, 0.f), (float)(H - 1));
}

// ---- BorderAlign: bilinear sample 11 points per border, max over points ----
__global__ void border_align(const h16* __restrict__ lt, const float* __restrict__ boxes,
                             h16* __restrict__ al, int H, int W, int Cb) {
  const int pk = blockIdx.x;
  const int p = pk >> 2, k = pk & 3;
  const int c = threadIdx.x;
  const int C4 = Cb * 4;
  const float x1 = boxes[p * 4 + 0], y1 = boxes[p * 4 + 1];
  const float x2 = boxes[p * 4 + 2], y2 = boxes[p * 4 + 3];
  const int ch = k * Cb + c;
  float vmax = -3.4e38f;
  for (int t = 0; t <= 10; ++t) {
    float ft = (float)t / 10.f;
    float x, y;
    if (k == 0)      { x = x1 + (x2 - x1) * ft; y = y1; }
    else if (k == 1) { x = x1; y = y1 + (y2 - y1) * ft; }
    else if (k == 2) { x = x1 + (x2 - x1) * ft; y = y2; }
    else             { x = x2; y = y1 + (y2 - y1) * ft; }
    float x0f = floorf(x), y0f = floorf(y);
    float lx = x - x0f, ly = y - y0f;
    int x0 = min(max((int)x0f, 0), W - 1);
    int x1i = min(x0 + 1, W - 1);
    int y0 = min(max((int)y0f, 0), H - 1);
    int y1i = min(y0 + 1, H - 1);
    float v00 = (float)lt[(long)(y0 * W + x0) * C4 + ch];
    float v01 = (float)lt[(long)(y0 * W + x1i) * C4 + ch];
    float v10 = (float)lt[(long)(y1i * W + x0) * C4 + ch];
    float v11 = (float)lt[(long)(y1i * W + x1i) * C4 + ch];
    float v = v00 * (1.f - ly) * (1.f - lx) + v01 * (1.f - ly) * lx +
              v10 * ly * (1.f - lx) + v11 * ly * lx;
    vmax = fmaxf(vmax, v);
  }
  al[(long)p * C4 + k * Cb + c] = (h16)vmax;
}

// ---- concat([al(transposed+reshaped), fm_short]) into NHWC f16 -------------
__global__ void build_cat(const h16* __restrict__ al, const h16* __restrict__ fm,
                          h16* __restrict__ cat, int M, int Cb, int Cs) {
  const int C4 = 4 * Cb, Ctot = C4 + Cs;
  long i = (long)blockIdx.x * blockDim.x + threadIdx.x;
  if (i >= (long)M * Ctot) return;
  int q = (int)(i % Ctot);
  int r = (int)(i / Ctot);
  h16 v;
  if (q < C4) {
    // reference: al (N,HW,4,C) -> transpose(0,3,1,2) -> reshape(N,4C,H,W)
    long f = (long)q * M + r;
    int c = (int)(f / (4L * M));
    long rem = f % (4L * M);
    int p = (int)(rem >> 2);
    int k = (int)(rem & 3);
    v = al[(long)p * C4 + k * Cb + c];
  } else {
    v = fm[(long)r * Cs + (q - C4)];
  }
  cat[i] = v;
}

// ===========================================================================
extern "C" void kernel_launch(void* const* d_in, const int* in_sizes, int n_in,
                              void* d_out, int out_size, void* d_ws, size_t ws_size,
                              hipStream_t stream) {
  (void)in_sizes; (void)n_in; (void)out_size; (void)ws_size;
  const float* featp[3]  = {(const float*)d_in[0], (const float*)d_in[2], (const float*)d_in[4]};
  const float* shiftp[3] = {(const float*)d_in[1], (const float*)d_in[3], (const float*)d_in[5]};
  const float* cls_tw = (const float*)d_in[6];
  const float* cls_tb = (const float*)d_in[7];
  const float* cls_gng = (const float*)d_in[8];
  const float* cls_gnb = (const float*)d_in[9];
  const float* box_tw = (const float*)d_in[10];
  const float* box_tb = (const float*)d_in[11];
  const float* box_gng = (const float*)d_in[12];
  const float* box_gnb = (const float*)d_in[13];
  const float* cls_logits_w = (const float*)d_in[14];
  const float* cls_logits_b = (const float*)d_in[15];
  const float* bbox_w = (const float*)d_in[16];
  const float* bbox_b = (const float*)d_in[17];
  const float* ctr_w = (const float*)d_in[18];
  const float* ctr_b = (const float*)d_in[19];
  const float* scales = (const float*)d_in[20];
  const float* bc_cur_w = (const float*)d_in[21];
  const float* bc_cur_b = (const float*)d_in[22];
  const float* bc_ltrb_w = (const float*)d_in[23];
  const float* bc_ltrb_b = (const float*)d_in[24];
  const float* bc_out_w = (const float*)d_in[25];
  const float* bc_out_b = (const float*)d_in[26];
  const float* bc_score_w = (const float*)d_in[27];
  const float* bc_score_b = (const float*)d_in[28];
  const float* bb_cur_w = (const float*)d_in[29];
  const float* bb_cur_b = (const float*)d_in[30];
  const float* bb_ltrb_w = (const float*)d_in[31];
  const float* bb_ltrb_b = (const float*)d_in[32];
  const float* bb_out_w = (const float*)d_in[33];
  const float* bb_out_b = (const float*)d_in[34];
  const float* bb_pred_w = (const float*)d_in[35];
  const float* bb_pred_b = (const float*)d_in[36];
  float* out = (float*)d_out;

  const int Hs[3] = {64, 32, 16};
  const float strd[3] = {8.f, 16.f, 32.f};
  const long HWs[3] = {4096, 1024, 256};
  const int C = 256, K3 = 2304;  // 9*256

  // ---- d_out offsets (return order) ----
  long off = 0, off_log[3], off_bbx[3], off_ctr[3], off_blg[3], off_bbr[3], off_pre[3];
  for (int l = 0; l < 3; ++l) { off_log[l] = off; off += 80 * HWs[l]; }
  for (int l = 0; l < 3; ++l) { off_bbx[l] = off; off += 4 * HWs[l]; }
  for (int l = 0; l < 3; ++l) { off_ctr[l] = off; off += HWs[l]; }
  for (int l = 0; l < 3; ++l) { off_blg[l] = off; off += 80 * HWs[l]; }
  for (int l = 0; l < 3; ++l) { off_bbr[l] = off; off += 4 * HWs[l]; }
  for (int l = 0; l < 3; ++l) { off_pre[l] = off; off += 4 * HWs[l]; }

  // ---- workspace bump allocator ----
  char* wp = (char*)d_ws;
  auto grab = [&wp](size_t bytes) -> char* {
    char* r = wp;
    wp += (bytes + 255) & ~(size_t)255;
    return r;
  };
  h16* zpage  = (h16*)grab(4096L * 2);  // zeroed pad page
  h16* clsTp  = (h16*)grab(4L * 256 * K3 * 2);
  h16* boxTp  = (h16*)grab(4L * 256 * K3 * 2);
  h16* logitsp = (h16*)grab(80L * K3 * 2);
  h16* bboxp  = (h16*)grab(16L * K3 * 2);
  h16* ctrp   = (h16*)grab(16L * K3 * 2);
  h16* bccurp = (h16*)grab(256L * 256 * 2);
  h16* bcltrbp = (h16*)grab(1024L * 256 * 2);
  h16* bcoutp = (h16*)grab(256L * 1280 * 2);
  h16* bcscorep = (h16*)grab(80L * 256 * 2);
  h16* bbcurp = (h16*)grab(128L * 256 * 2);
  h16* bbltrbp = (h16*)grab(512L * 256 * 2);
  h16* bboutp = (h16*)grab(256L * 640 * 2);
  h16* bbpredp = (h16*)grab(16L * 256 * 2);
  h16* featH  = (h16*)grab(4096L * 256 * 2);
  h16* cA = (h16*)grab(4096L * 256 * 2);
  h16* cB = (h16*)grab(4096L * 256 * 2);
  h16* bA = (h16*)grab(4096L * 256 * 2);
  h16* bB = (h16*)grab(4096L * 256 * 2);
  float* convf = (float*)grab(4096L * 1024 * 4);
  h16* fmb  = (h16*)grab(4096L * 256 * 2);
  h16* ltb  = (h16*)grab(4096L * 1024 * 2);
  h16* alb  = (h16*)grab(4096L * 1024 * 2);
  h16* catb = (h16*)grab(4096L * 1280 * 2);
  h16* brout = (h16*)grab(4096L * 256 * 2);
  float* boxesb = (float*)grab(4096L * 4 * 4);
  float* gstats = (float*)grab(64 * 4);
  float* istats = (float*)grab(2048 * 4);

  zero_h16<<<16, 256, 0, stream>>>(zpage, 4096);

  auto pack = [&](const float* w_, h16* bt, int O, int Npad, int Cin, int khw) {
    long tot = (long)Npad * khw * Cin;
    pack_w<<<(int)((tot + 255) / 256), 256, 0, stream>>>(w_, bt, O, Npad, Cin, khw);
  };
  auto gemm = [&](const h16* A, const h16* Bt, const float* bias,
                  const float* alpha_ptr, float beta, float* oF, h16* oH,
                  int M, int Npad, int Nreal, int taps, int H, int W,
                  int Cin, int epi) {
    dim3 grid(M / 64, (Npad + 127) / 128);
    dim3 block(32, 4);
    gemm_wmma<<<grid, block, 0, stream>>>(A, Bt, zpage, bias, alpha_ptr, beta,
                                          oF, oH, M, Npad, Nreal, taps, H, W,
                                          Cin, epi);
  };

  // ---- pack all weights (shared across levels) ----
  for (int i = 0; i < 4; ++i) {
    pack(cls_tw + (long)i * C * C * 9, clsTp + (long)i * 256 * K3, 256, 256, C, 9);
    pack(box_tw + (long)i * C * C * 9, boxTp + (long)i * 256 * K3, 256, 256, C, 9);
  }
  pack(cls_logits_w, logitsp, 80, 80, C, 9);
  pack(bbox_w, bboxp, 4, 16, C, 9);
  pack(ctr_w, ctrp, 1, 16, C, 9);
  pack(bc_cur_w, bccurp, 256, 256, C, 1);
  pack(bc_ltrb_w, bcltrbp, 1024, 1024, C, 1);
  pack(bc_out_w, bcoutp, 256, 256, 1280, 1);
  pack(bc_score_w, bcscorep, 80, 80, C, 1);
  pack(bb_cur_w, bbcurp, 128, 128, C, 1);
  pack(bb_ltrb_w, bbltrbp, 512, 512, C, 1);
  pack(bb_out_w, bboutp, 256, 256, 640, 1);
  pack(bb_pred_w, bbpredp, 4, 16, C, 1);

  auto run_border = [&](const h16* xt, const h16* curp, const float* cur_b, int Ccur,
                        const h16* ltrbp, const float* ltrb_b, int Cltrb,
                        const h16* outp, const float* out_b,
                        const h16* predp, const float* pred_b, int Npred, int NpadPred,
                        float* dst, int HW, int H, int W) {
    const int Cb = Cltrb / 4, Ctot = Cltrb + Ccur;
    // fm_short = relu(inorm(1x1 conv))
    gemm(xt, curp, cur_b, nullptr, 0.f, convf, nullptr, HW, Ccur, Ccur, 1, H, W, C, 0);
    in_stats<<<Ccur, 256, 0, stream>>>(convf, istats, Ccur, HW);
    in_apply<<<(int)(((long)HW * Ccur + 255) / 256), 256, 0, stream>>>(convf, istats, fmb, Ccur, HW);
    // lt = relu(inorm(1x1 conv))
    gemm(xt, ltrbp, ltrb_b, nullptr, 0.f, convf, nullptr, HW, Cltrb, Cltrb, 1, H, W, C, 0);
    in_stats<<<Cltrb, 256, 0, stream>>>(convf, istats, Cltrb, HW);
    in_apply<<<(int)(((long)HW * Cltrb + 255) / 256), 256, 0, stream>>>(convf, istats, ltb, Cltrb, HW);
    // border align + concat
    border_align<<<HW * 4, Cb, 0, stream>>>(ltb, boxesb, alb, H, W, Cb);
    build_cat<<<(int)(((long)HW * Ctot + 255) / 256), 256, 0, stream>>>(alb, fmb, catb, HW, Cb, Ccur);
    // out conv (relu, f16 NHWC), then pred conv -> NCHW f32 into d_out
    gemm(catb, outp, out_b, nullptr, 0.f, nullptr, brout, HW, 256, 256, 1, H, W, Ctot, 3);
    gemm(brout, predp, pred_b, nullptr, 0.f, dst, nullptr, HW, NpadPred, Npred, 1, H, W, C, 1);
  };

  for (int l = 0; l < 3; ++l) {
    const int H = Hs[l], W = Hs[l];
    const int HW = (int)HWs[l];
    // feat NCHW f32 -> NHWC f16
    nchw_to_nhwc_h<<<(int)(((long)C * HW + 255) / 256), 256, 0, stream>>>(featp[l], featH, C, HW);

    // cls tower
    const h16* x = featH;
    h16* cbuf[2] = {cA, cB};
    for (int i = 0; i < 4; ++i) {
      gemm(x, clsTp + (long)i * 256 * K3, cls_tb + i * C, nullptr, 0.f, convf, nullptr,
           HW, 256, 256, 9, H, W, C, 0);
      gn_stats<<<32, 256, 0, stream>>>(convf, gstats, HW);
      gn_apply<<<(int)(((long)HW * 256 + 255) / 256), 256, 0, stream>>>(
          convf, gstats, cls_gng + i * C, cls_gnb + i * C, cbuf[i & 1], HW);
      x = cbuf[i & 1];
    }
    const h16* cls_t = x;

    // box tower
    x = featH;
    h16* bbuf[2] = {bA, bB};
    for (int i = 0; i < 4; ++i) {
      gemm(x, boxTp + (long)i * 256 * K3, box_tb + i * C, nullptr, 0.f, convf, nullptr,
           HW, 256, 256, 9, H, W, C, 0);
      gn_stats<<<32, 256, 0, stream>>>(convf, gstats, HW);
      gn_apply<<<(int)(((long)HW * 256 + 255) / 256), 256, 0, stream>>>(
          convf, gstats, box_gng + i * C, box_gnb + i * C, bbuf[i & 1], HW);
      x = bbuf[i & 1];
    }
    const h16* box_t = x;

    // heads: logits, ctr, bbox_reg = relu(scales[l]*conv)*stride
    gemm(cls_t, logitsp, cls_logits_b, nullptr, 0.f, out + off_log[l], nullptr,
         HW, 80, 80, 9, H, W, C, 1);
    gemm(box_t, ctrp, ctr_b, nullptr, 0.f, out + off_ctr[l], nullptr,
         HW, 16, 1, 9, H, W, C, 1);
    gemm(box_t, bboxp, bbox_b, scales + l, strd[l], out + off_bbx[l], nullptr,
         HW, 16, 4, 9, H, W, C, 2);

    // pre_bbox + clipped boxes
    make_boxes<<<(HW + 255) / 256, 256, 0, stream>>>(
        out + off_bbx[l], shiftp[l], out + off_pre[l], boxesb, HW, H, W, strd[l]);

    // border branches
    run_border(cls_t, bccurp, bc_cur_b, 256, bcltrbp, bc_ltrb_b, 1024,
               bcoutp, bc_out_b, bcscorep, bc_score_b, 80, 80,
               out + off_blg[l], HW, H, W);
    run_border(box_t, bbcurp, bb_cur_b, 128, bbltrbp, bb_ltrb_b, 512,
               bboutp, bb_out_b, bbpredp, bb_pred_b, 4, 16,
               out + off_bbr[l], HW, H, W);
  }
}
